// RNN_19258633355834
// MI455X (gfx1250) — matrix operand order repaired and checked by reading, not compiled, not run on hardware
//
#include <hip/hip_runtime.h>
#include <hip/hip_bf16.h>
#include <math.h>

// ---------------------------------------------------------------------------
// Recurrent MLP on MI455X (gfx1250, wave32, WMMA).
//
// 65536 strictly sequential steps; per step three GEMVs (512x512, 512x512,
// 512x256) + tanh/tanh/sigmoid, output feeds back. Single persistent
// workgroup (32 waves on one WGP). Each 16-output tile = chain of 16
// v_wmma_f32_16x16x32_f16 over K=512. A = prepacked W^T fragments; B carries
// the activation vector replicated into all 16 columns (no zeroing, no lane
// divergence; D is replicated across N, we read column 0).
//
// Data movement: W1+W2 (1 MB f16) streamed from the 192 MB L2 each step
// (regular RT/device-scope loads; LICM across the step loop is defeated by
// laundering the fragment index through an empty asm). W3 (256 KB f16) is
// staged once into LDS (320 KB/WGP) and served by ds_load_b128. Fragment
// loads are double-buffered so each WMMA waits with >=2 loads in flight,
// and the next step's x chunk is prefetched under layers 2-3.
// ---------------------------------------------------------------------------

typedef __attribute__((ext_vector_type(16))) _Float16 v16h;
typedef __attribute__((ext_vector_type(8)))  float    v8f;

#define T_STEPS 65536
#define RAW     256
#define H1D     512
#define H2D     512
#define OUTD    256

// ---- prepack weights into WMMA A-fragment order ---------------------------
// 16-bit A-matrix 16x32 layout (ISA 7.12.2):
//   lanes 0-15 : M=lane,    slots hold K = {0..7, 16..23}
//   lanes 16-31: M=lane-16, slots hold K = {8..15, 24..31}
// dst element = ((tile*16 + kstep)*32 + lane)*16 + slot  -> 32 B/lane chunks.
__global__ void pack_w_kernel(const float* __restrict__ w,
                              _Float16* __restrict__ wp,
                              int K, int N) {
    int total = K * N;
    for (int e = blockIdx.x * blockDim.x + threadIdx.x; e < total;
         e += gridDim.x * blockDim.x) {
        int s  = e & 15;          // slot within lane
        int L  = (e >> 4) & 31;   // lane
        int t  = (e >> 9) & 15;   // k-step (K chunk of 32)
        int tt = e >> 13;         // output tile (16 neurons)
        int m  = L & 15;
        int klocal;
        if (L < 16) klocal = (s < 8) ? s     : s + 8;   // 0..7, 16..23
        else        klocal = (s < 8) ? s + 8 : s + 16;  // 8..15, 24..31
        int k = t * 32 + klocal;
        int n = tt * 16 + m;
        wp[e] = (_Float16)w[k * N + n];   // A[m][k] = W[k][n]
    }
}

__global__ void cvt_f16_kernel(const float* __restrict__ src,
                               _Float16* __restrict__ dst, int n) {
    for (int i = blockIdx.x * blockDim.x + threadIdx.x; i < n;
         i += gridDim.x * blockDim.x)
        dst[i] = (_Float16)src[i];
}

// ---- fast saturating activations -----------------------------------------
__device__ __forceinline__ float fast_tanh(float x) {
    // 1 - 2/(1+e^{2x}) : correct saturation at +/-inf (no NaN on overflow)
    return 1.0f - 2.0f / (1.0f + __expf(2.0f * x));
}
__device__ __forceinline__ float fast_sigmoid(float x) {
    return 1.0f / (1.0f + __expf(-x));
}

// ---- one 16-output GEMV tile: 16 chained WMMAs over K=512 -----------------
// B-matrix 32x16 (16-bit): lane n (0-15) holds column n, K={0..7,16..23}; lanes
// 16-31 hold column n-16, K={8..15,24..31}. Every lane loads the h chunk, so all 16
// columns equal h -> D replicated across N, column 0 is the GEMV result.
//
// Double-buffered: the t+1 fragment loads are issued before WMMA t, so the
// wait before each WMMA keeps 2 loads in flight instead of draining to 0.
__device__ __forceinline__ v8f wmma_gemv512(const _Float16* w,
                                            const _Float16* h_lds,
                                            int tile, int lane) {
    const v16h* W = (const v16h*)w;
    int idx0 = tile * 16 * 32 + lane;       // v16h units
    asm volatile("" : "+v"(idx0));          // opaque each step: defeat LICM
    const int khalf = (lane >> 4) * 8;      // 0 for lanes 0-15, 8 for 16-31
    typedef __attribute__((ext_vector_type(8))) _Float16 v8h_;
    v8f c = {};
    v16h a_cur = W[idx0];
    v16h b_cur = __builtin_shufflevector(*(const v8h_*)(h_lds + khalf), *(const v8h_*)(h_lds + 16 + khalf), 0,1,2,3,4,5,6,7,8,9,10,11,12,13,14,15);
#pragma unroll
    for (int t = 0; t < 16; ++t) {
        v16h a_nxt = a_cur;
        v16h b_nxt = b_cur;
        if (t < 15) {                       // compile-time after full unroll
            a_nxt = W[idx0 + (t + 1) * 32];
            b_nxt = __builtin_shufflevector(*(const v8h_*)(h_lds + (t + 1) * 32 + khalf), *(const v8h_*)(h_lds + (t + 1) * 32 + 16 + khalf), 0,1,2,3,4,5,6,7,8,9,10,11,12,13,14,15);
        }
        c = __builtin_amdgcn_wmma_f32_16x16x32_f16(
                false, a_cur, false, b_cur, (short)0, c, false, false);
        a_cur = a_nxt;
        b_cur = b_nxt;
    }
    return c;
}

// ---- persistent recurrent kernel: one workgroup, 32 waves -----------------
__global__ __launch_bounds__(1024, 1)
void rnn_persistent(const _Float16* __restrict__ xp,
                    const _Float16* __restrict__ w1p,
                    const _Float16* __restrict__ w2p,
                    const _Float16* __restrict__ w3p,
                    const float* __restrict__ b1,
                    const float* __restrict__ b2,
                    const float* __restrict__ b3,
                    float* __restrict__ out) {
    // 256 KB W3 + 3 KB activations of static LDS (<= 320 KB per WGP)
    alignas(32) __shared__ _Float16 lds_w3[H2D * OUTD];
    alignas(32) __shared__ _Float16 hbuf[H1D];   // [x_t(256) | o_prev(256)]
    alignas(32) __shared__ _Float16 h1o[H1D];    // layer-1 output
    alignas(32) __shared__ _Float16 h2o[H2D];    // layer-2 output

    const int tid  = threadIdx.x;
    const int lane = tid & 31;
    const int wave = tid >> 5;

    // stage W3 fragments into LDS once (8192 x 32 B chunks, 8 per thread)
    {
        const v16h* src = (const v16h*)w3p;
        v16h*       dst = (v16h*)lds_w3;
#pragma unroll
        for (int i = 0; i < 8; ++i)
            dst[tid + i * 1024] = src[tid + i * 1024];
    }
    if (tid < OUTD) hbuf[RAW + tid] = (_Float16)0.0f;   // o_0 = 0

    // preload x chunk for step 0 into registers (16 lanes x 32 B)
    v16h xreg = {};
    if (tid < 16) xreg = ((const v16h*)xp)[tid];
    __syncthreads();

#pragma unroll 1
    for (int step = 0; step < T_STEPS; ++step) {
        // commit the prefetched x chunk for this step
        if (tid < 16) ((v16h*)hbuf)[tid] = xreg;
        __syncthreads();

        // ---- layer 1: 512 -> 512, tile = wave (32 tiles) ----
        {
            v8f c = wmma_gemv512(w1p, hbuf, wave, lane);

            // prefetch next step's x chunk; latency hidden under layers 2-3
            if (tid < 16) {
                int nxt = (step + 1 < T_STEPS) ? step + 1 : step;
                xreg = ((const v16h*)(xp + (size_t)nxt * RAW))[tid];
            }

            if ((lane & 15) == 0) {
                // D layout: VGPR r, lane 0 -> M=r ; lane 16 -> M=8+r
                int mbase = (lane >> 4) * 8;
#pragma unroll
                for (int i = 0; i < 8; ++i) {
                    int n = wave * 16 + mbase + i;
                    h1o[n] = (_Float16)fast_tanh(c[i] + b1[n]);
                }
            }
        }
        __syncthreads();

        // ---- layer 2: 512 -> 512 ----
        {
            v8f c = wmma_gemv512(w2p, h1o, wave, lane);
            if ((lane & 15) == 0) {
                int mbase = (lane >> 4) * 8;
#pragma unroll
                for (int i = 0; i < 8; ++i) {
                    int n = wave * 16 + mbase + i;
                    h2o[n] = (_Float16)fast_tanh(c[i] + b2[n]);
                }
            }
        }
        __syncthreads();

        // ---- layer 3: 512 -> 256, waves 0..15, W3 from LDS ----
        if (wave < 16) {
            v8f c = wmma_gemv512(lds_w3, h2o, wave, lane);
            if ((lane & 15) == 0) {
                int mbase = (lane >> 4) * 8;
#pragma unroll
                for (int i = 0; i < 8; ++i) {
                    int n = wave * 16 + mbase + i;
                    float o = fast_sigmoid(c[i] + b3[n]);
                    hbuf[RAW + n] = (_Float16)o;         // feedback
                    if (step == T_STEPS - 1) out[n] = o; // final result
                }
            }
        }
        __syncthreads();
    }
}

// ---------------------------------------------------------------------------
extern "C" void kernel_launch(void* const* d_in, const int* in_sizes, int n_in,
                              void* d_out, int out_size, void* d_ws, size_t ws_size,
                              hipStream_t stream) {
    (void)in_sizes; (void)n_in; (void)out_size; (void)ws_size;

    const float* x  = (const float*)d_in[0];   // [65536, 256]
    const float* W1 = (const float*)d_in[1];   // [512, 512]
    const float* b1 = (const float*)d_in[2];   // [512]
    const float* W2 = (const float*)d_in[3];   // [512, 512]
    const float* b2 = (const float*)d_in[4];   // [512]
    const float* W3 = (const float*)d_in[5];   // [512, 256]
    const float* b3 = (const float*)d_in[6];   // [256]
    float* out = (float*)d_out;                // [256]

    // workspace layout (f16 elements)
    _Float16* ws  = (_Float16*)d_ws;
    _Float16* w1p = ws;                         // 512*512
    _Float16* w2p = w1p + 512 * 512;            // 512*512
    _Float16* w3p = w2p + 512 * 512;            // 512*256
    _Float16* xp  = w3p + 512 * 256;            // 65536*256

    pack_w_kernel<<<512, 512, 0, stream>>>(W1, w1p, 512, 512);
    pack_w_kernel<<<512, 512, 0, stream>>>(W2, w2p, 512, 512);
    pack_w_kernel<<<256, 512, 0, stream>>>(W3, w3p, 512, 256);
    cvt_f16_kernel<<<2048, 512, 0, stream>>>(x, xp, T_STEPS * RAW);

    rnn_persistent<<<1, 1024, 0, stream>>>(xp, w1p, w2p, w3p, b1, b2, b3, out);
}
